// GCN_9517647528031
// MI455X (gfx1250) — matrix operand order repaired and checked by reading, not compiled, not run on hardware
//
#include <hip/hip_runtime.h>

// ---------------------------------------------------------------------------
// GCN (two GCNConv layers + meanmax readout + linear head) for MI455X gfx1250
// GEMMs use v_wmma_f32_16x16x32_bf16 (bf16 in, fp32 accumulate).
// Weights pre-converted/transposed to bf16 once; B tiles staged to LDS with
// global_load_async_to_lds_b128 (ASYNCcnt); A tiles converted in-flight with
// native bf16 converts. Edge scatter uses hardware global_atomic_add_f32
// (both feature matrices are L2-resident: 105MB each vs 192MB L2).
// ---------------------------------------------------------------------------

typedef __attribute__((ext_vector_type(16))) __bf16 v16bf;
typedef __attribute__((ext_vector_type(8)))  __bf16 v8bf;
typedef __attribute__((ext_vector_type(8)))  float  v8f;

#define GCN_N   102400
#define GCN_F   256
#define GCN_K1  400
#define GCN_E   4096000
#define GCN_G   256
#define GCN_NPG 400     // nodes per graph

#define BM 128
#define BN 128
#define BK 32
#define LDK 40          // padded LDS row stride (elements) -> 80B, 16B aligned

#define KP1 448         // K1=400 padded to multiple of 64 (bf16 row stride)
#define KP2 256         // K2=256 already aligned

// ---------------------------------------------------------------------------
// Degree pipeline: deg[i] = 1 (self loop) + #edges with dst==i;  dis = rsqrt
// ---------------------------------------------------------------------------
__global__ void deg_init(float* __restrict__ deg) {
  int i = blockIdx.x * blockDim.x + threadIdx.x;
  if (i < GCN_N) deg[i] = 1.0f;
}

__global__ void deg_edges(const int* __restrict__ dst, float* __restrict__ deg) {
  int e = blockIdx.x * blockDim.x + threadIdx.x;
  if (e < GCN_E)
    __hip_atomic_fetch_add(&deg[dst[e]], 1.0f, __ATOMIC_RELAXED,
                           __HIP_MEMORY_SCOPE_AGENT);
}

__global__ void deg_rsqrt(float* __restrict__ deg) {
  int i = blockIdx.x * blockDim.x + threadIdx.x;
  if (i < GCN_N) deg[i] = rsqrtf(deg[i]);
}

// ---------------------------------------------------------------------------
// Pre-convert weights: Wt[n][kp] = bf16(W[k][n]), zero-padded to Kp.
// Tiny one-time pass (<=400KB); makes GEMM B-staging pure async b128 DMA.
// ---------------------------------------------------------------------------
__global__ void wt_build(const float* __restrict__ W, __bf16* __restrict__ Wt,
                         int K, int Kp) {
  int idx = blockIdx.x * blockDim.x + threadIdx.x;   // over 256*Kp
  int n = idx / Kp;
  int k = idx - n * Kp;
  float v = (k < K) ? W[(size_t)k * GCN_F + n] : 0.0f;
  Wt[idx] = (__bf16)v;
}

// ---------------------------------------------------------------------------
// C[M,N] = act(A[M,K]) @ W[K,N]   via v_wmma_f32_16x16x32_bf16
// A fp32 row-major (converted in-flight); Wt bf16 pre-transposed [n][Kp].
// Block tile 128x128, K-step 32, 8 waves -> each wave 32x64 of C.
// ---------------------------------------------------------------------------
__global__ __launch_bounds__(256)
void gemm_bf16_wmma(const float* __restrict__ A, const __bf16* __restrict__ Wt,
                    float* __restrict__ C, int M, int K, int Kp, int N,
                    int relu_a) {
  __shared__ __align__(16) __bf16 As[BM * LDK];      // [m][k]
  __shared__ __align__(16) __bf16 Bs[BN * LDK];      // [n][k]

  const int tid  = threadIdx.x;
  const int lane = tid & 31;
  const int wave = tid >> 5;         // 0..7
  const int wm   = wave & 3;         // M tile row of wave  (wm*32)
  const int wn   = wave >> 2;        // N tile col of wave  (wn*64)
  const int half = lane >> 4;        // lane group 0/1
  const int l16  = lane & 15;

  const int bm0 = blockIdx.x * BM;
  const int bn0 = blockIdx.y * BN;

  v8f acc[2][4];
  #pragma unroll
  for (int i = 0; i < 2; ++i)
    #pragma unroll
    for (int j = 0; j < 4; ++j) { v8f z = {}; acc[i][j] = z; }

  // global->LDS staging assignments (each thread: one 16-elem run per matrix)
  const int amrow = tid >> 1;          // 0..127
  const int akoff = (tid & 1) * 16;    // 0 or 16
  const int bnrow = tid >> 1;          // 0..127
  const int bkoff = (tid & 1) * 16;    // 0 or 16

  const float*  abase = A  + (size_t)(bm0 + amrow) * K  + akoff;
  const __bf16* bbase = Wt + (size_t)(bn0 + bnrow) * Kp + bkoff;
  v8bf* adst = (v8bf*)&As[amrow * LDK + akoff];
  // LDS byte offset for async DMA destination (low 32 bits of shared address)
  const unsigned ldsB = (unsigned)(unsigned long long)&Bs[bnrow * LDK + bkoff];

  const int ksteps = (K + BK - 1) / BK;
  for (int kt = 0; kt < ksteps; ++kt) {
    const int k0 = kt * BK;

    // --- stage B tile: async global->LDS DMA (32B/thread), overlaps A work ---
    {
      const __bf16* bsrc = bbase + k0;
      asm volatile(
          "global_load_async_to_lds_b128 %0, %1, off\n\t"
          "global_load_async_to_lds_b128 %0, %1, off offset:16"
          :: "v"(ldsB), "v"(bsrc) : "memory");
    }

    // --- stage A tile (fp32 -> bf16), run-uniform bounds guard (K%16==0) ---
    {
      const float* asrc = abase + k0;
      if (kt + 1 < ksteps) __builtin_prefetch(asrc + BK, 0, 1);
      v8bf lo, hi;
      if (k0 + akoff < K) {
        const float4* s4 = (const float4*)asrc;
        float4 f0 = s4[0], f1 = s4[1], f2 = s4[2], f3 = s4[3];
        if (relu_a) {
          f0.x = fmaxf(f0.x, 0.f); f0.y = fmaxf(f0.y, 0.f);
          f0.z = fmaxf(f0.z, 0.f); f0.w = fmaxf(f0.w, 0.f);
          f1.x = fmaxf(f1.x, 0.f); f1.y = fmaxf(f1.y, 0.f);
          f1.z = fmaxf(f1.z, 0.f); f1.w = fmaxf(f1.w, 0.f);
          f2.x = fmaxf(f2.x, 0.f); f2.y = fmaxf(f2.y, 0.f);
          f2.z = fmaxf(f2.z, 0.f); f2.w = fmaxf(f2.w, 0.f);
          f3.x = fmaxf(f3.x, 0.f); f3.y = fmaxf(f3.y, 0.f);
          f3.z = fmaxf(f3.z, 0.f); f3.w = fmaxf(f3.w, 0.f);
        }
        lo[0] = (__bf16)f0.x; lo[1] = (__bf16)f0.y; lo[2] = (__bf16)f0.z; lo[3] = (__bf16)f0.w;
        lo[4] = (__bf16)f1.x; lo[5] = (__bf16)f1.y; lo[6] = (__bf16)f1.z; lo[7] = (__bf16)f1.w;
        hi[0] = (__bf16)f2.x; hi[1] = (__bf16)f2.y; hi[2] = (__bf16)f2.z; hi[3] = (__bf16)f2.w;
        hi[4] = (__bf16)f3.x; hi[5] = (__bf16)f3.y; hi[6] = (__bf16)f3.z; hi[7] = (__bf16)f3.w;
      } else {
        #pragma unroll
        for (int i = 0; i < 8; ++i) { lo[i] = (__bf16)0.0f; hi[i] = (__bf16)0.0f; }
      }
      adst[0] = lo;
      adst[1] = hi;
    }

    // wait for this wave's async DMA, then block barrier
    asm volatile("s_wait_asynccnt 0" ::: "memory");
    __syncthreads();

    // --- build fragments per ISA 7.12.2 layouts ---
    // A 16x32 bf16: lane l16 -> M row; VGPR v<4: K = half*8+2v(+1); v>=4: +16
    // B 32x16 bf16 (stored [n][k]): lane l16 -> N col; K same as A (half*8+[0..7], +16)
    v16bf afr[2], bfr[4];
    #pragma unroll
    for (int t = 0; t < 2; ++t) {
      const __bf16* ap = &As[(wm * 32 + t * 16 + l16) * LDK + half * 8];
      #pragma unroll
      for (int i = 0; i < 8; ++i) { afr[t][i] = ap[i]; afr[t][8 + i] = ap[16 + i]; }
    }
    #pragma unroll
    for (int t = 0; t < 4; ++t) {
      const __bf16* bp = &Bs[(wn * 64 + t * 16 + l16) * LDK + half * 8];
      #pragma unroll
      for (int i = 0; i < 8; ++i) { bfr[t][i] = bp[i]; bfr[t][8 + i] = bp[16 + i]; }
    }

    #pragma unroll
    for (int i = 0; i < 2; ++i)
      #pragma unroll
      for (int j = 0; j < 4; ++j)
        acc[i][j] = __builtin_amdgcn_wmma_f32_16x16x32_bf16(
            false, afr[i], false, bfr[j], (short)0, acc[i][j], false, false);

    __syncthreads();
  }

  // --- write C: 16x16 f32 D layout: VGPR r -> M = r + half*8, lane l16 -> N ---
  #pragma unroll
  for (int i = 0; i < 2; ++i) {
    #pragma unroll
    for (int j = 0; j < 4; ++j) {
      const int mbase = bm0 + wm * 32 + i * 16 + half * 8;
      const int nn    = bn0 + wn * 64 + j * 16 + l16;
      #pragma unroll
      for (int r = 0; r < 8; ++r)
        C[(size_t)(mbase + r) * N + nn] = acc[i][j][r];
    }
  }
}

// ---------------------------------------------------------------------------
// out[n][f] = h[n][f]*dis[n]^2 + b[f]   (self-loop term + bias, also zero-init)
// ---------------------------------------------------------------------------
__global__ __launch_bounds__(256)
void scatter_self_bias(const float* __restrict__ h, const float* __restrict__ dis,
                       const float* __restrict__ b, float* __restrict__ out) {
  size_t idx = (size_t)blockIdx.x * blockDim.x + threadIdx.x;
  int n = (int)(idx >> 8);        // F == 256
  int f = (int)(idx & 255);
  float di = dis[n];
  out[idx] = h[idx] * di * di + b[f];
}

// ---------------------------------------------------------------------------
// For each edge: out[dst] += h[src] * dis[src]*dis[dst].  One wave per edge,
// 8 floats per lane, hardware f32 atomics (L2-resident working set).
// ---------------------------------------------------------------------------
__global__ __launch_bounds__(256)
void scatter_edges(const int* __restrict__ src_idx, const int* __restrict__ dst_idx,
                   const float* __restrict__ h, const float* __restrict__ dis,
                   float* __restrict__ out) {
  const int e = blockIdx.x * 8 + (threadIdx.x >> 5);
  const int lane = threadIdx.x & 31;
  const int s = src_idx[e];
  const int d = dst_idx[e];
  const float norm = dis[s] * dis[d];
  const float4* hp = (const float4*)(h + (size_t)s * GCN_F) + lane * 2;
  float* op = out + (size_t)d * GCN_F + lane * 8;
  float4 v0 = hp[0];
  float4 v1 = hp[1];
  float vals[8] = {v0.x, v0.y, v0.z, v0.w, v1.x, v1.y, v1.z, v1.w};
  #pragma unroll
  for (int i = 0; i < 8; ++i)
    __hip_atomic_fetch_add(&op[i], vals[i] * norm, __ATOMIC_RELAXED,
                           __HIP_MEMORY_SCOPE_AGENT);
}

// ---------------------------------------------------------------------------
// meanmax readout per graph: g[gr][0:256]=mean, g[gr][256:512]=max
// ---------------------------------------------------------------------------
__global__ __launch_bounds__(256)
void readout(const float* __restrict__ h, float* __restrict__ g) {
  const int gr = blockIdx.x;
  const int f  = threadIdx.x;
  const float* p = h + (size_t)gr * GCN_NPG * GCN_F + f;
  float s = 0.0f, mx = -__builtin_inff();
  for (int n = 0; n < GCN_NPG; ++n) {
    float v = p[(size_t)n * GCN_F];
    s += v;
    mx = fmaxf(mx, v);
  }
  g[gr * 512 + f]       = s * (1.0f / (float)GCN_NPG);
  g[gr * 512 + 256 + f] = mx;
}

// ---------------------------------------------------------------------------
// out[gr][j] = sum_k g[gr][k] * Wm[k][j] + bm[j], j in {0,1}. One wave/graph.
// ---------------------------------------------------------------------------
__global__ __launch_bounds__(32)
void final_mlp(const float* __restrict__ g, const float* __restrict__ Wm,
               const float* __restrict__ bm, float* __restrict__ out) {
  const int gr = blockIdx.x;
  const int lane = threadIdx.x;
  float a0 = 0.0f, a1 = 0.0f;
  for (int k = lane; k < 512; k += 32) {
    float v = g[gr * 512 + k];
    a0 += v * Wm[k * 2 + 0];
    a1 += v * Wm[k * 2 + 1];
  }
  #pragma unroll
  for (int sft = 16; sft > 0; sft >>= 1) {
    a0 += __shfl_down(a0, sft, 32);
    a1 += __shfl_down(a1, sft, 32);
  }
  if (lane == 0) {
    out[gr * 2 + 0] = a0 + bm[0];
    out[gr * 2 + 1] = a1 + bm[1];
  }
}

// ---------------------------------------------------------------------------
extern "C" void kernel_launch(void* const* d_in, const int* in_sizes, int n_in,
                              void* d_out, int out_size, void* d_ws, size_t ws_size,
                              hipStream_t stream) {
  (void)in_sizes; (void)n_in; (void)out_size; (void)ws_size;

  const float* x  = (const float*)d_in[0];
  const int*   ei = (const int*)d_in[1];      // [2, E] -> src row, dst row
  // d_in[2] (batch) and d_in[3] (num_graphs) are implied by fixed geometry
  const float* W1 = (const float*)d_in[4];
  const float* b1 = (const float*)d_in[5];
  const float* W2 = (const float*)d_in[6];
  const float* b2 = (const float*)d_in[7];
  const float* Wm = (const float*)d_in[8];
  const float* bm = (const float*)d_in[9];
  float* out = (float*)d_out;

  const int* esrc = ei;
  const int* edst = ei + GCN_E;

  // workspace layout
  char* ws = (char*)d_ws;
  size_t off = 0;
  float*  dis  = (float*)(ws + off);  off += (size_t)GCN_N * 4;          // 0.41 MB
  float*  gbuf = (float*)(ws + off);  off += (size_t)GCN_G * 512 * 4;    // 0.5 MB
  __bf16* Wt1  = (__bf16*)(ws + off); off += (size_t)GCN_F * KP1 * 2;    // 224 KB
  __bf16* Wt2  = (__bf16*)(ws + off); off += (size_t)GCN_F * KP2 * 2;    // 128 KB
  off = (off + 255) & ~(size_t)255;
  float*  bufA = (float*)(ws + off);  off += (size_t)GCN_N * GCN_F * 4;  // 105 MB
  float*  bufB = (float*)(ws + off);                                     // 105 MB

  // 1) degrees -> deg^{-1/2}; pre-transposed bf16 weights
  deg_init  <<<GCN_N / 256, 256, 0, stream>>>(dis);
  deg_edges <<<(GCN_E + 255) / 256, 256, 0, stream>>>(edst, dis);
  deg_rsqrt <<<GCN_N / 256, 256, 0, stream>>>(dis);
  wt_build  <<<(GCN_F * KP1) / 256, 256, 0, stream>>>(W1, Wt1, GCN_K1, KP1);
  wt_build  <<<(GCN_F * KP2) / 256, 256, 0, stream>>>(W2, Wt2, GCN_F, KP2);

  dim3 gemm_grid(GCN_N / BM, GCN_F / BN);

  // 2) GEMM1: h1 = x @ W1                        (bufA)
  gemm_bf16_wmma<<<gemm_grid, 256, 0, stream>>>(x, Wt1, bufA,
                                                GCN_N, GCN_K1, KP1, GCN_F, 0);

  // 3) conv1 scatter: out1 = D^-1/2 A D^-1/2 h1 + b1   (bufB; ReLU folded into GEMM2)
  scatter_self_bias<<<GCN_N, 256, 0, stream>>>(bufA, dis, b1, bufB);
  scatter_edges    <<<GCN_E / 8, 256, 0, stream>>>(esrc, edst, bufA, dis, bufB);

  // 4) GEMM2: h2 = relu(out1) @ W2               (bufA)
  gemm_bf16_wmma<<<gemm_grid, 256, 0, stream>>>(bufB, Wt2, bufA,
                                                GCN_N, GCN_F, KP2, GCN_F, 1);

  // 5) conv2 scatter: out2 = D^-1/2 A D^-1/2 h2 + b2   (bufB)
  scatter_self_bias<<<GCN_N, 256, 0, stream>>>(bufA, dis, b2, bufB);
  scatter_edges    <<<GCN_E / 8, 256, 0, stream>>>(esrc, edst, bufA, dis, bufB);

  // 6) meanmax readout + head
  readout  <<<GCN_G, 256, 0, stream>>>(bufB, gbuf);
  final_mlp<<<GCN_G, 32, 0, stream>>>(gbuf, Wm, bm, out);
}
